// AggregatedBilinear_1726576856961
// MI455X (gfx1250) — matrix operand order repaired and checked by reading, not compiled, not run on hardware
//
#include <hip/hip_runtime.h>
#include <hip/hip_bf16.h>
#include <stdint.h>

typedef __bf16 bf16_t;
typedef bf16_t v16bf __attribute__((ext_vector_type(16)));
typedef float  v8f   __attribute__((ext_vector_type(8)));
typedef float  v4f   __attribute__((ext_vector_type(4)));   // native vector for NT loads
typedef uint32_t v8u __attribute__((ext_vector_type(8)));

#define B_TOTAL   65536
#define F_IN      512
#define I_FEAT    128
#define OUT_F     512
#define TM        128      // batch rows per block
#define NTHREADS  256      // 8 waves (wave32)
#define KSTRIDE   36       // bf16 elems per staged 32-wide row (+4 pad)
#define XSTRIDE   132      // bf16 elems per 128-wide row (+4 pad)

__device__ __forceinline__ float elu_f(float x) {
    return x > 0.f ? x : (__expf(x) - 1.f);
}

__global__ __launch_bounds__(NTHREADS)
void fused_agg_bilinear_kernel(const float* __restrict__ in1,
                               const float* __restrict__ in2,
                               const float* __restrict__ Wi1,
                               const float* __restrict__ Wi2,
                               const float* __restrict__ Wb,
                               const float* __restrict__ Wo,
                               float* __restrict__ out)
{
    // LDS carve: 3 * (128x132 bf16) + 8KB f32 Wbil + 2 staging tiles = 125 KB
    __shared__ __align__(16) unsigned char smem[
        3 * TM * XSTRIDE * 2 + 2048 * 4 + TM * KSTRIDE * 2 + I_FEAT * KSTRIDE * 2];

    bf16_t* sX1 = (bf16_t*)smem;
    bf16_t* sX2 = sX1 + TM * XSTRIDE;
    bf16_t* sY  = sX2 + TM * XSTRIDE;
    float*  sWb = (float*)(smem + 3 * TM * XSTRIDE * 2);
    bf16_t* sA  = (bf16_t*)((unsigned char*)sWb + 2048 * 4);
    bf16_t* sB  = sA + TM * KSTRIDE;
    bf16_t* sWo = sX1;   // reused in phase C (x1 dead by then)

    const int tid   = threadIdx.x;
    const int lane  = tid & 31;
    const int halfw = lane >> 4;   // 0/1 (lane 0-15 vs 16-31)
    const int l15   = lane & 15;
    const int wave  = tid >> 5;    // 0..7
    const int mrow  = wave * 16;   // local M-strip base
    const int row0  = blockIdx.x * TM;

    // fixed per-thread staging coordinates (TM x 32 tile as 1024 float4s)
    const int sr[4] = { (tid + 0)   >> 3, (tid + 256) >> 3,
                        (tid + 512) >> 3, (tid + 768) >> 3 };
    const int sc = (tid & 7) * 4;  // column-of-float4 start (same for all i)

    // preload bilinear weights (32*4*4*4 = 2048 floats; L2-resident)
    #pragma unroll
    for (int i = 0; i < 8; ++i)
        sWb[tid + NTHREADS * i] = Wb[tid + NTHREADS * i];

    // ============ Phase A: X{1,2} = ELU(in @ W_in^T), bf16 WMMA ============
    for (int p = 0; p < 2; ++p) {
        const float* inp = p ? in2 : in1;
        const float* W   = p ? Wi2 : Wi1;
        bf16_t* sX = p ? sX2 : sX1;

        v8f acc[8];
        #pragma unroll
        for (int nt = 0; nt < 8; ++nt)
            acc[nt] = (v8f){0.f,0.f,0.f,0.f,0.f,0.f,0.f,0.f};

        // register double-buffer: prefetch chunk kc=0
        v4f pA[4], pB[4];
        #pragma unroll
        for (int i = 0; i < 4; ++i) {
            // streaming, read-once -> nontemporal (don't evict L2 weights)
            pA[i] = __builtin_nontemporal_load(
                (const v4f*)(inp + (size_t)(row0 + sr[i]) * F_IN + sc));
            pB[i] = *(const v4f*)(W + (size_t)sr[i] * F_IN + sc);
        }

        for (int kc = 0; kc < F_IN; kc += 32) {
            __syncthreads();   // previous iter's fragment reads done
            // commit prefetched tiles to LDS as bf16
            #pragma unroll
            for (int i = 0; i < 4; ++i) {
                bf16_t* da = sA + sr[i] * KSTRIDE + sc;
                da[0] = (bf16_t)pA[i].x; da[1] = (bf16_t)pA[i].y;
                da[2] = (bf16_t)pA[i].z; da[3] = (bf16_t)pA[i].w;
                bf16_t* db = sB + sr[i] * KSTRIDE + sc;
                db[0] = (bf16_t)pB[i].x; db[1] = (bf16_t)pB[i].y;
                db[2] = (bf16_t)pB[i].z; db[3] = (bf16_t)pB[i].w;
            }
            __syncthreads();

            // issue next chunk's global loads; latency hidden by WMMAs below
            int kn = kc + 32;
            if (kn < F_IN) {
                #pragma unroll
                for (int i = 0; i < 4; ++i) {
                    pA[i] = __builtin_nontemporal_load(
                        (const v4f*)(inp + (size_t)(row0 + sr[i]) * F_IN + kn + sc));
                    pB[i] = *(const v4f*)(W + (size_t)sr[i] * F_IN + kn + sc);
                }
            }

            // A fragment: 16-bit A 16x32 layout (packed bf16 pairs, dword LDS loads)
            v8u au;
            #pragma unroll
            for (int v = 0; v < 8; ++v) {
                int kb = ((v & 4) ? 16 : 0) + halfw * 8 + 2 * (v & 3);
                au[v] = *(const uint32_t*)(sA + (mrow + l15) * KSTRIDE + kb);
            }
            v16bf a = __builtin_bit_cast(v16bf, au);

            #pragma unroll
            for (int nt = 0; nt < 8; ++nt) {
                v8u bu;
                #pragma unroll
                for (int v = 0; v < 8; ++v) {
                    int kb = ((v & 4) ? 16 : 0) + halfw * 8 + 2 * (v & 3);
                    bu[v] = *(const uint32_t*)(sB + (nt * 16 + l15) * KSTRIDE + kb);
                }
                v16bf b = __builtin_bit_cast(v16bf, bu);
                acc[nt] = __builtin_amdgcn_wmma_f32_16x16x32_bf16(
                    false, a, false, b, (short)0, acc[nt], false, false);
            }
        }
        __syncthreads();
        // ELU + store X strip (C/D layout: VGPR j -> row halfw*8+j, col = lane&15)
        #pragma unroll
        for (int nt = 0; nt < 8; ++nt) {
            #pragma unroll
            for (int j = 0; j < 8; ++j) {
                float e = elu_f(acc[nt][j]);
                sX[(mrow + halfw * 8 + j) * XSTRIDE + nt * 16 + l15] = (bf16_t)e;
            }
        }
    }
    __syncthreads();

    // ============ Phase B: per-group bilinear + ELU (VALU) ============
    {
        int r  = tid >> 1;           // 128 rows over 256 threads
        int c0 = (tid & 1) * 16;     // 16 groups each
        for (int c = c0; c < c0 + 16; ++c) {
            float a4[4], b4[4];
            #pragma unroll
            for (int i = 0; i < 4; ++i) {
                a4[i] = (float)sX1[r * XSTRIDE + c * 4 + i];
                b4[i] = (float)sX2[r * XSTRIDE + c * 4 + i];
            }
            const float* w = sWb + c * 64;   // [o][i][j]
            #pragma unroll
            for (int o = 0; o < 4; ++o) {
                float y = 0.f;
                #pragma unroll
                for (int i = 0; i < 4; ++i) {
                    float s = 0.f;
                    #pragma unroll
                    for (int j = 0; j < 4; ++j)
                        s = fmaf(w[o * 16 + i * 4 + j], b4[j], s);
                    y = fmaf(a4[i], s, y);
                }
                sY[r * XSTRIDE + c * 4 + o] = (bf16_t)elu_f(y);
            }
        }
    }
    __syncthreads();

    // ============ Phase C: out = Y @ W_out^T, 4 column chunks of 128 ============
    for (int nc = 0; nc < 4; ++nc) {
        __syncthreads();
        // stage W_out chunk [128 n][128 k] bf16 (weights are L2-resident)
        #pragma unroll
        for (int i = 0; i < 16; ++i) {
            int idx = tid + NTHREADS * i;    // 4096 float4s
            int n = idx >> 5, c4 = idx & 31;
            v4f f = *(const v4f*)(Wo + (size_t)(nc * 128 + n) * I_FEAT + c4 * 4);
            bf16_t* dst = sWo + n * XSTRIDE + c4 * 4;
            dst[0] = (bf16_t)f.x; dst[1] = (bf16_t)f.y;
            dst[2] = (bf16_t)f.z; dst[3] = (bf16_t)f.w;
        }
        __syncthreads();

        v8f acc[8];
        #pragma unroll
        for (int nt = 0; nt < 8; ++nt)
            acc[nt] = (v8f){0.f,0.f,0.f,0.f,0.f,0.f,0.f,0.f};

        #pragma unroll
        for (int kc = 0; kc < I_FEAT; kc += 32) {
            v8u au;
            #pragma unroll
            for (int v = 0; v < 8; ++v) {
                int kb = kc + ((v & 4) ? 16 : 0) + halfw * 8 + 2 * (v & 3);
                au[v] = *(const uint32_t*)(sY + (mrow + l15) * XSTRIDE + kb);
            }
            v16bf a = __builtin_bit_cast(v16bf, au);
            #pragma unroll
            for (int nt = 0; nt < 8; ++nt) {
                v8u bu;
                #pragma unroll
                for (int v = 0; v < 8; ++v) {
                    int kb = kc + ((v & 4) ? 16 : 0) + halfw * 8 + 2 * (v & 3);
                    bu[v] = *(const uint32_t*)(sWo + (nt * 16 + l15) * XSTRIDE + kb);
                }
                v16bf b = __builtin_bit_cast(v16bf, bu);
                acc[nt] = __builtin_amdgcn_wmma_f32_16x16x32_bf16(
                    false, a, false, b, (short)0, acc[nt], false, false);
            }
        }
        // write-once results -> nontemporal stores (lanes 0-15: 64B bursts)
        #pragma unroll
        for (int nt = 0; nt < 8; ++nt) {
            #pragma unroll
            for (int j = 0; j < 8; ++j) {
                int grow = row0 + mrow + halfw * 8 + j;
                int gcol = nc * 128 + nt * 16 + l15;
                __builtin_nontemporal_store(acc[nt][j],
                                            out + (size_t)grow * OUT_F + gcol);
            }
        }
    }
}

extern "C" void kernel_launch(void* const* d_in, const int* in_sizes, int n_in,
                              void* d_out, int out_size, void* d_ws, size_t ws_size,
                              hipStream_t stream) {
    const float* in1 = (const float*)d_in[0];
    const float* in2 = (const float*)d_in[1];
    const float* Wi1 = (const float*)d_in[2];
    const float* Wi2 = (const float*)d_in[3];
    const float* Wb  = (const float*)d_in[4];
    const float* Wo  = (const float*)d_in[5];
    float* outp = (float*)d_out;

    dim3 grid(B_TOTAL / TM);
    dim3 block(NTHREADS);
    hipLaunchKernelGGL(fused_agg_bilinear_kernel, grid, block, 0, stream,
                       in1, in2, Wi1, Wi2, Wb, Wo, outp);
}
